// AttentionHead_4776003633411
// MI455X (gfx1250) — matrix operand order repaired and checked by reading, not compiled, not run on hardware
//
#include <hip/hip_runtime.h>

typedef __attribute__((ext_vector_type(16))) _Float16 v16h;
typedef __attribute__((ext_vector_type(8)))  _Float16 v8h;
typedef __attribute__((ext_vector_type(2)))  _Float16 h2;
typedef __attribute__((ext_vector_type(8)))  float    v8f;
typedef __attribute__((ext_vector_type(4)))  unsigned u32x4;
typedef __attribute__((ext_vector_type(4)))  int      i32x4;
typedef __attribute__((ext_vector_type(8)))  int      i32x8;

#define D_MODEL 1024
#define DK      64
#define SEQ     2048
#define NB      4

static __device__ __forceinline__ unsigned pack2(float a, float b) {
  h2 p;
  p.x = (_Float16)a;
  p.y = (_Float16)b;
  return __builtin_bit_cast(unsigned, p);
}

static __device__ __forceinline__ v8f wmma_f16(v16h a, v16h b, v8f c) {
  return __builtin_amdgcn_wmma_f32_16x16x32_f16(false, a, false, b, (short)0, c,
                                                false, false);
}

// TDM: 1-D copy of 8KB (1024 x 8-byte elements) global -> LDS.
// D# per CDNA5 ISA: count=1, type=2,
// data_size=3 (8B), tensor_dim0=tile_dim0=1024, dims1..4 = 1 (in-bounds).
static __device__ __forceinline__ void tdm_load_8kb(const unsigned* src,
                                                    unsigned* lds_dst) {
  unsigned long long ga = (unsigned long long)(uintptr_t)src;
  unsigned lds_off = (unsigned)(uintptr_t)lds_dst;  // low 32b of flat = LDS offset
  const unsigned N = 1024;  // 8-byte elements

  u32x4 g0;
  g0.x = 1u;                                                // count = 1
  g0.y = lds_off;                                           // lds_addr
  g0.z = (unsigned)ga;                                      // global_addr[31:0]
  g0.w = (unsigned)((ga >> 32) & 0x01FFFFFFu) | 0x80000000u;// ga[56:32] | type=2

  i32x8 g1;
  g1[0] = (int)(3u << 16);                 // workgroup_mask=0, data_size=8B
  g1[1] = (int)((N & 0xFFFFu) << 16);      // tensor_dim0[15:0]
  g1[2] = (int)((N >> 16) | (1u << 16));   // tensor_dim0[31:16] | tensor_dim1[15:0]
  g1[3] = (int)(N << 16);                  // tensor_dim1[31:16]=0 | tile_dim0
  g1[4] = 1;                               // tile_dim1=1, tile_dim2=0(unused)
  g1[5] = (int)N;                          // tensor_dim0_stride[31:0]
  g1[6] = (int)((N & 0xFFFFu) << 16);      // stride0[47:32]=0 | stride1[15:0]
  g1[7] = 0;                               // stride1[47:16]

  i32x4 g2 = {1, 1, 0, 0};                 // tensor_dim2=1, tensor_dim3=1
  i32x4 g3 = {0, (int)(1u << 16), 0, 0};   // tensor_dim4=1, tile_dim4=0

#if defined(__clang_major__) && (__clang_major__ >= 23)
  i32x8 gz = {};
  __builtin_amdgcn_tensor_load_to_lds(g0, g1, g2, g3, gz, 0);
#else
  __builtin_amdgcn_tensor_load_to_lds(g0, g1, g2, g3, 0);
#endif
}

// ---------------------------------------------------------------------------
// Projection: acc = X[128 rows] * W + bias, then repack f16 register images:
//   mode 0 (Q): A-layout tiles, pre-scaled by 1/sqrt(64)  -> AQ
//   mode 1 (K): B-layout (K^T, f16 pairs along d)         -> KVP K-sections
//   mode 2 (V): B-layout (V, f16 pairs along key)         -> KVP V-sections
// ---------------------------------------------------------------------------
__global__ __launch_bounds__(256) void proj_kernel(
    const float* __restrict__ X, const float* __restrict__ W,
    const float* __restrict__ bias, unsigned* __restrict__ pout, int mode) {
  __shared__ __align__(16) unsigned wbuf[4][32][8];  // W slice, B register image
  __shared__ float sbuf[128][68];                    // fp32 results (C->pack bounce)

  const int tid = threadIdx.x;
  const int w   = tid >> 5;
  const int l   = tid & 31;
  const int h   = l >> 4;
  const int ln  = l & 15;
  const int rowbase = blockIdx.x * 128 + w * 16;
  const int arow    = rowbase + ln;

  v8f acc[4] = {};

  for (int mb = 0; mb < D_MODEL; mb += 32) {
    __syncthreads();
    {
      int idx = tid * 4;  // 4*32*8 = 1024 dwords
#pragma unroll
      for (int q = 0; q < 4; ++q, ++idx) {
        int nt   = idx >> 8;
        int lane = (idx >> 3) & 31;
        int v    = idx & 7;
        int m0   = mb + ((lane >> 4) << 3) + ((v < 4) ? 2 * v : 16 + 2 * (v - 4));
        int n    = (nt << 4) + (lane & 15);
        wbuf[nt][lane][v] = pack2(W[(size_t)m0 * DK + n],
                                  W[(size_t)(m0 + 1) * DK + n]);
      }
    }
    __syncthreads();

    v16h a;
    {
      const float* xp = X + (size_t)arow * D_MODEL + mb;
      float4 x0 = *(const float4*)(xp + 8 * h);
      float4 x1 = *(const float4*)(xp + 8 * h + 4);
      float4 x2 = *(const float4*)(xp + 16 + 8 * h);
      float4 x3 = *(const float4*)(xp + 16 + 8 * h + 4);
      a[0]  = (_Float16)x0.x;  a[1]  = (_Float16)x0.y;
      a[2]  = (_Float16)x0.z;  a[3]  = (_Float16)x0.w;
      a[4]  = (_Float16)x1.x;  a[5]  = (_Float16)x1.y;
      a[6]  = (_Float16)x1.z;  a[7]  = (_Float16)x1.w;
      a[8]  = (_Float16)x2.x;  a[9]  = (_Float16)x2.y;
      a[10] = (_Float16)x2.z;  a[11] = (_Float16)x2.w;
      a[12] = (_Float16)x3.x;  a[13] = (_Float16)x3.y;
      a[14] = (_Float16)x3.z;  a[15] = (_Float16)x3.w;
    }

#pragma unroll
    for (int nt = 0; nt < 4; ++nt) {
      v16h b = *(const v16h*)&wbuf[nt][l][0];
      acc[nt] = wmma_f16(a, b, acc[nt]);
    }
  }

  // C + bias into LDS (C layout: VGPR i -> row i+8h, col = lane%16).
#pragma unroll
  for (int nt = 0; nt < 4; ++nt) {
    int   n  = (nt << 4) + ln;
    float bv = bias[n];
#pragma unroll
    for (int i = 0; i < 8; ++i)
      sbuf[w * 16 + i + 8 * h][n] = acc[nt][i] + bv;
  }
  __syncthreads();

  if (mode == 0) {
    // AQ: per 16-row tile, [lane][16 dwords] A-register image, scaled by 1/8.
    unsigned* base = pout + (size_t)blockIdx.x * 4096;
#pragma unroll
    for (int q = 0; q < 16; ++q) {
      int idx  = q * 256 + tid;       // 8*512 dwords
      int tl   = idx >> 9;
      int lane = (idx >> 4) & 31;
      int vv   = idx & 15;
      int hh   = lane >> 4;
      int at   = vv >> 3;
      int vp   = vv & 7;
      int d0   = 32 * at + ((vp < 4) ? (8 * hh + 2 * vp)
                                     : (16 + 8 * hh + 2 * (vp - 4)));
      int lr   = tl * 16 + (lane & 15);
      base[idx] = pack2(sbuf[lr][d0] * 0.125f, sbuf[lr][d0 + 1] * 0.125f);
    }
  } else if (mode == 1) {
    int b   = blockIdx.x >> 4;
    int jt0 = (blockIdx.x & 15) * 8;  // 16-key tile base within batch
    unsigned* kvb = pout + (size_t)b * 64 * 2048;
#pragma unroll
    for (int q = 0; q < 16; ++q) {
      int idx  = q * 256 + tid;       // 8*512 dwords
      int tl   = idx >> 9;
      int rem  = idx & 511;
      int dc   = rem >> 8;
      int lane = (rem >> 3) & 31;
      int v    = rem & 7;
      int hh   = lane >> 4;
      int keyl = tl * 16 + (lane & 15);
      int d0   = dc * 32 + ((v < 4) ? (8 * hh + 2 * v) : (16 + 8 * hh + 2 * (v - 4)));
      int jt   = jt0 + tl;
      size_t off = (size_t)(jt >> 1) * 2048 + (size_t)(jt & 1) * 512 +
                   dc * 256 + lane * 8 + v;
      kvb[off] = pack2(sbuf[keyl][d0], sbuf[keyl][d0 + 1]);
    }
  } else {
    int b   = blockIdx.x >> 4;
    int jv0 = (blockIdx.x & 15) * 4;  // 32-key tile base within batch
    unsigned* kvb = pout + (size_t)b * 64 * 2048;
#pragma unroll
    for (int q = 0; q < 16; ++q) {
      int idx  = q * 256 + tid;       // 4*1024 dwords
      int jl   = idx >> 10;
      int rem  = idx & 1023;
      int t    = rem >> 8;
      int lane = (rem >> 3) & 31;
      int v    = rem & 7;
      int hh   = lane >> 4;
      int k0   = jl * 32 + ((v < 4) ? (8 * hh + 2 * v) : (16 + 8 * hh + 2 * (v - 4)));
      int n    = t * 16 + (lane & 15);
      size_t off = (size_t)(jv0 + jl) * 2048 + 1024 + t * 256 + lane * 8 + v;
      kvb[off] = pack2(sbuf[k0][n], sbuf[k0 + 1][n]);
    }
  }
}

// ---------------------------------------------------------------------------
// Flash attention: WG = 128 queries of one batch. K/V tiles arrive via
// double-buffered TDM tensor_load_to_lds; inner loop is WMMA + softmax only.
// ---------------------------------------------------------------------------
__global__ __launch_bounds__(256) void attn_kernel(
    const unsigned* __restrict__ AQ, const unsigned* __restrict__ KVP,
    float* __restrict__ Out) {
  __shared__ __align__(16) unsigned kvbuf[2][2048];   // 2 x 8KB K+V tiles
  __shared__ __align__(16) _Float16 pbuf[8][16][32];  // per-wave P C->A bounce

  const int b   = blockIdx.y;
  const int tid = threadIdx.x;
  const int w   = tid >> 5;
  const int l   = tid & 31;
  const int h   = l >> 4;
  const int ln  = l & 15;
  const int qbase = blockIdx.x * 128 + w * 16;

  const unsigned* kvsrc = KVP + (size_t)b * 64 * 2048;

  // Q A-tiles: prepacked register image, 16 dwords/lane.
  v16h aq0, aq1;
  {
    const v16h* qp = (const v16h*)(AQ +
        ((size_t)(b * 128 + blockIdx.x * 8 + w)) * 512 + (size_t)l * 16);
    aq0 = qp[0];
    aq1 = qp[1];
  }

  v8f   o[4] = {};
  float m_i[8], l_i[8];
#pragma unroll
  for (int i = 0; i < 8; ++i) { m_i[i] = -1e30f; l_i[i] = 0.0f; }

  if (w == 0) tdm_load_8kb(kvsrc, &kvbuf[0][0]);

  for (int jv = 0; jv < 64; ++jv) {
    if (w == 0) {
      if (jv + 1 < 64) {
        tdm_load_8kb(kvsrc + (size_t)(jv + 1) * 2048, &kvbuf[(jv + 1) & 1][0]);
        __builtin_amdgcn_s_wait_tensorcnt(1);  // oldest (current tile) done
      } else {
        __builtin_amdgcn_s_wait_tensorcnt(0);
      }
    }
    __syncthreads();  // publish kvbuf[jv&1] to all waves
    const unsigned* kv = &kvbuf[jv & 1][0];

    // Scores: two 16x16 C tiles over d = 2 K-chunks.
    v8f s0 = {}, s1 = {};
    s0 = wmma_f16(aq0, *(const v16h*)(kv + 0 * 256 + l * 8), s0);
    s0 = wmma_f16(aq1, *(const v16h*)(kv + 1 * 256 + l * 8), s0);
    s1 = wmma_f16(aq0, *(const v16h*)(kv + 2 * 256 + l * 8), s1);
    s1 = wmma_f16(aq1, *(const v16h*)(kv + 3 * 256 + l * 8), s1);

    // Online softmax; row r = i + 8h lives in VGPR i across this lane-half.
    float corr[8];
#pragma unroll
    for (int i = 0; i < 8; ++i) {
      float mx = fmaxf(s0[i], s1[i]);
#pragma unroll
      for (int mk = 1; mk < 16; mk <<= 1)
        mx = fmaxf(mx, __shfl_xor(mx, mk, 32));
      float mn = fmaxf(m_i[i], mx);
      corr[i] = __expf(m_i[i] - mn);
      m_i[i] = mn;
      float p0 = __expf(s0[i] - mn);
      float p1 = __expf(s1[i] - mn);
      s0[i] = p0;
      s1[i] = p1;
      float rs = p0 + p1;
#pragma unroll
      for (int mk = 1; mk < 16; mk <<= 1)
        rs += __shfl_xor(rs, mk, 32);
      l_i[i] = l_i[i] * corr[i] + rs;
#pragma unroll
      for (int t = 0; t < 4; ++t) o[t][i] *= corr[i];
    }

    // P: C layout -> A layout via per-wave LDS bounce.
#pragma unroll
    for (int i = 0; i < 8; ++i) {
      int r = i + 8 * h;
      pbuf[w][r][ln]      = (_Float16)s0[i];
      pbuf[w][r][16 + ln] = (_Float16)s1[i];
    }
    v16h ap;
    {
      v8h p0 = *(const v8h*)&pbuf[w][ln][8 * h];
      v8h p1 = *(const v8h*)&pbuf[w][ln][16 + 8 * h];
      ap = __builtin_shufflevector(p0, p1, 0, 1, 2, 3, 4, 5, 6, 7, 8, 9, 10, 11,
                                   12, 13, 14, 15);
    }

    // O += P * V (K = 32 keys).
#pragma unroll
    for (int t = 0; t < 4; ++t)
      o[t] = wmma_f16(ap, *(const v16h*)(kv + 1024 + t * 256 + l * 8), o[t]);

    __syncthreads();  // all reads of kvbuf[jv&1] done before its next refill
  }

  // Epilogue: divide by row sums, store fp32.
#pragma unroll
  for (int i = 0; i < 8; ++i) {
    float inv = 1.0f / l_i[i];
    int   row = qbase + i + 8 * h;
#pragma unroll
    for (int t = 0; t < 4; ++t) {
      int n = (t << 4) + ln;
      Out[((size_t)b * SEQ + row) * DK + n] = o[t][i] * inv;
    }
  }
}

// ---------------------------------------------------------------------------
extern "C" void kernel_launch(void* const* d_in, const int* in_sizes, int n_in,
                              void* d_out, int out_size, void* d_ws,
                              size_t ws_size, hipStream_t stream) {
  const float* query = (const float*)d_in[0];
  const float* key   = (const float*)d_in[1];
  const float* value = (const float*)d_in[2];
  const float* Wq    = (const float*)d_in[3];
  const float* bq    = (const float*)d_in[4];
  const float* Wk    = (const float*)d_in[5];
  const float* bk    = (const float*)d_in[6];
  const float* Wv    = (const float*)d_in[7];
  const float* bv    = (const float*)d_in[8];
  float*       out   = (float*)d_out;

  unsigned* AQ  = (unsigned*)d_ws;            // 512 tiles * 512 dwords = 1 MB
  unsigned* KVP = AQ + (size_t)512 * 512;     // 256 tiles * 2048 dwords = 2 MB

  dim3 pgrid((NB * SEQ) / 128, 1);
  proj_kernel<<<pgrid, 256, 0, stream>>>(query, Wq, bq, AQ, 0);
  proj_kernel<<<pgrid, 256, 0, stream>>>(key, Wk, bk, KVP, 1);
  proj_kernel<<<pgrid, 256, 0, stream>>>(value, Wv, bv, KVP, 2);

  dim3 agrid(SEQ / 128, NB);
  attn_kernel<<<agrid, 256, 0, stream>>>(AQ, KVP, out);
}
